// FrameTransformerGenerator_70849780515275
// MI455X (gfx1250) — matrix operand order repaired and checked by reading, not compiled, not run on hardware
//
#include <hip/hip_runtime.h>

// ---------------- constants (match reference) ----------------
constexpr int Bn = 2, Cn = 4, Hn = 4, Fn = 1024, Wn = 1024, HD = 256, CE = 16, Ln = 3;
constexpr long long FW = (long long)Fn * Wn;   // 1M
constexpr long long WW = (long long)Wn * Wn;   // 1M

typedef __attribute__((ext_vector_type(16))) __bf16        bf16x16;
typedef __attribute__((ext_vector_type(8)))  float         f32x8;
typedef __attribute__((ext_vector_type(4)))  unsigned int  u32x4;

union FragU { bf16x16 v; u32x4 q[2]; };

__device__ __forceinline__ unsigned short f2bf(float f) {
  unsigned u = __float_as_uint(f);
  unsigned r = 0x7FFFu + ((u >> 16) & 1u);
  return (unsigned short)((u + r) >> 16);
}

// ---------------- elementwise f32 -> bf16 ----------------
__global__ __launch_bounds__(256) void f2bf_kernel(const float* __restrict__ in,
                                                   unsigned short* __restrict__ out,
                                                   long long n) {
  long long t = (long long)blockIdx.x * blockDim.x + threadIdx.x;
  if (t < n) out[t] = f2bf(in[t]);
}

// ---------------- conv3x3 (C=4 -> 1 map) + transpose + bf16 ----------------
// xt[b][w][f] = conv(hin)[b,0,f,w]
__global__ __launch_bounds__(256) void conv_qkv_bt_kernel(
    const float* __restrict__ x, const float* __restrict__ wgt,
    const float* __restrict__ bias, unsigned short* __restrict__ xt) {
  long long t = (long long)blockIdx.x * blockDim.x + threadIdx.x;
  if (t >= (long long)Bn * FW) return;
  int w = (int)(t & (Wn - 1));
  long long r = t >> 10;
  int f = (int)(r & (Fn - 1));
  int b = (int)(r >> 10);
  float acc = bias[0];
#pragma unroll
  for (int ci = 0; ci < Cn; ++ci) {
    const float* xp = x + (long long)(b * Cn + ci) * FW;
    const float* wp = wgt + ci * 9;
#pragma unroll
    for (int df = -1; df <= 1; ++df) {
      int ff = f + df;
      if (ff < 0 || ff >= Fn) continue;
#pragma unroll
      for (int dw = -1; dw <= 1; ++dw) {
        int ww = w + dw;
        if (ww < 0 || ww >= Wn) continue;
        acc += xp[(long long)ff * Wn + ww] * wp[(df + 1) * 3 + (dw + 1)];
      }
    }
  }
  xt[((long long)b * Wn + w) * Fn + f] = f2bf(acc);
}

// ---------------- generic conv3x3 (runtime Cin/Cout, optional exact gelu) ----------------
__global__ __launch_bounds__(256) void conv3x3_kernel(
    const float* __restrict__ in, const float* __restrict__ wgt,
    const float* __restrict__ bias, float* __restrict__ out,
    int Cin, int Cout, int do_gelu) {
  long long t = (long long)blockIdx.x * blockDim.x + threadIdx.x;
  long long total = (long long)Bn * Cout * FW;
  if (t >= total) return;
  int w = (int)(t & (Wn - 1));
  long long r = t >> 10;
  int f = (int)(r & (Fn - 1));
  long long r2 = r >> 10;
  int co = (int)(r2 % Cout);
  int b = (int)(r2 / Cout);
  float acc = bias[co];
  for (int ci = 0; ci < Cin; ++ci) {
    const float* xp = in + (long long)(b * Cin + ci) * FW;
    const float* wp = wgt + (long long)(co * Cin + ci) * 9;
#pragma unroll
    for (int df = -1; df <= 1; ++df) {
      int ff = f + df;
      if (ff < 0 || ff >= Fn) continue;
#pragma unroll
      for (int dw = -1; dw <= 1; ++dw) {
        int ww = w + dw;
        if (ww < 0 || ww >= Wn) continue;
        acc += xp[(long long)ff * Wn + ww] * wp[(df + 1) * 3 + (dw + 1)];
      }
    }
  }
  if (do_gelu) acc = 0.5f * acc * (1.0f + erff(acc * 0.70710678118654752f));
  out[t] = acc;
}

// ---------------- o_proj conv: channels [hin(4), dw*olin(1)] -> 4 ----------------
__global__ __launch_bounds__(256) void oproj_conv_kernel(
    const float* __restrict__ hin, const float* __restrict__ olin,
    const float* __restrict__ dw, const float* __restrict__ wgt,
    const float* __restrict__ bias, float* __restrict__ out) {
  long long t = (long long)blockIdx.x * blockDim.x + threadIdx.x;
  if (t >= (long long)Bn * Cn * FW) return;
  int w = (int)(t & (Wn - 1));
  long long r = t >> 10;
  int f = (int)(r & (Fn - 1));
  long long r2 = r >> 10;
  int co = (int)(r2 & (Cn - 1));
  int b = (int)(r2 >> 2);
  float scale = dw[0];
  float acc = bias[co];
#pragma unroll
  for (int ci = 0; ci < Cn; ++ci) {
    const float* xp = hin + (long long)(b * Cn + ci) * FW;
    const float* wp = wgt + (long long)(co * (Cn + 1) + ci) * 9;
#pragma unroll
    for (int df = -1; df <= 1; ++df) {
      int ff = f + df;
      if (ff < 0 || ff >= Fn) continue;
#pragma unroll
      for (int dwi = -1; dwi <= 1; ++dwi) {
        int ww = w + dwi;
        if (ww < 0 || ww >= Wn) continue;
        acc += xp[(long long)ff * Wn + ww] * wp[(df + 1) * 3 + (dwi + 1)];
      }
    }
  }
  {
    const float* ap = olin + (long long)b * FW;
    const float* wp = wgt + (long long)(co * (Cn + 1) + Cn) * 9;
#pragma unroll
    for (int df = -1; df <= 1; ++df) {
      int ff = f + df;
      if (ff < 0 || ff >= Fn) continue;
#pragma unroll
      for (int dwi = -1; dwi <= 1; ++dwi) {
        int ww = w + dwi;
        if (ww < 0 || ww >= Wn) continue;
        acc += scale * ap[(long long)ff * Wn + ww] * wp[(df + 1) * 3 + (dwi + 1)];
      }
    }
  }
  out[t] = acc;
}

// ---------------- residual add + multichannel LayerNorm over F axis ----------------
__global__ __launch_bounds__(256) void add_ln_kernel(
    const float* __restrict__ x, const float* __restrict__ z,
    const float* __restrict__ gw, const float* __restrict__ gb,
    float* __restrict__ out) {
  int col = blockIdx.x;  // (b*Cn + c)*Wn + w
  int w = col & (Wn - 1);
  int bc = col >> 10;
  int c = bc & (Cn - 1);
  long long base = (long long)bc * FW + w;
  __shared__ float red[256];
  float vals[4];
  float s = 0.f;
#pragma unroll
  for (int i = 0; i < 4; ++i) {
    int f = threadIdx.x + i * 256;
    float v = x[base + (long long)f * Wn] + z[base + (long long)f * Wn];
    vals[i] = v;
    s += v;
  }
  red[threadIdx.x] = s;
  __syncthreads();
  for (int st = 128; st > 0; st >>= 1) {
    if ((int)threadIdx.x < st) red[threadIdx.x] += red[threadIdx.x + st];
    __syncthreads();
  }
  float mean = red[0] * (1.0f / Fn);
  __syncthreads();
  float vs = 0.f;
#pragma unroll
  for (int i = 0; i < 4; ++i) {
    float d = vals[i] - mean;
    vs += d * d;
  }
  red[threadIdx.x] = vs;
  __syncthreads();
  for (int st = 128; st > 0; st >>= 1) {
    if ((int)threadIdx.x < st) red[threadIdx.x] += red[threadIdx.x + st];
    __syncthreads();
  }
  float rstd = rsqrtf(red[0] * (1.0f / Fn) + 1e-5f);
#pragma unroll
  for (int i = 0; i < 4; ++i) {
    int f = threadIdx.x + i * 256;
    out[base + (long long)f * Wn] = (vals[i] - mean) * rstd * gw[c * Fn + f] + gb[c * Fn + f];
  }
}

// ---------------- rotary + head pack: Qh[b][h][w][dd] bf16 <- lin[b][g][w] f32 ----------------
__global__ __launch_bounds__(256) void rotary_pack_kernel(
    const float* __restrict__ lin, unsigned short* __restrict__ out) {
  long long t = (long long)blockIdx.x * blockDim.x + threadIdx.x;
  if (t >= (long long)Bn * Hn * Wn * HD) return;
  int dd = (int)(t & (HD - 1));
  long long r = t >> 8;
  int w = (int)(r & (Wn - 1));
  long long bh = r >> 10;
  int h = (int)(bh & (Hn - 1));
  int b = (int)(bh >> 2);
  int i = dd & 127;
  float inv = __expf(-(float)(2 * i) * (9.210340371976184f / 256.0f));
  float ang = (float)w * inv;
  float sn, cs;
  __sincosf(ang, &sn, &cs);
  long long gbase = ((long long)b * Fn + h * HD) * Wn;
  float xv = lin[gbase + (long long)dd * Wn + w];
  float pv = lin[gbase + (long long)(dd ^ 128) * Wn + w];
  float res = (dd < 128) ? (xv * cs - pv * sn) : (xv * cs + pv * sn);
  out[t] = f2bf(res);
}

// ---------------- row softmax (1024 cols) f32 -> bf16 probs ----------------
__global__ __launch_bounds__(256) void softmax_kernel(
    const float* __restrict__ qk, unsigned short* __restrict__ P) {
  long long row = blockIdx.x;
  const float* r = qk + row * Wn;
  __shared__ float red[256];
  float v[4];
  float mx = -3.4e38f;
#pragma unroll
  for (int i = 0; i < 4; ++i) {
    v[i] = r[threadIdx.x + i * 256];
    mx = fmaxf(mx, v[i]);
  }
  red[threadIdx.x] = mx;
  __syncthreads();
  for (int st = 128; st > 0; st >>= 1) {
    if ((int)threadIdx.x < st) red[threadIdx.x] = fmaxf(red[threadIdx.x], red[threadIdx.x + st]);
    __syncthreads();
  }
  float m = red[0];
  __syncthreads();
  float s = 0.f;
#pragma unroll
  for (int i = 0; i < 4; ++i) {
    v[i] = __expf(v[i] - m);
    s += v[i];
  }
  red[threadIdx.x] = s;
  __syncthreads();
  for (int st = 128; st > 0; st >>= 1) {
    if ((int)threadIdx.x < st) red[threadIdx.x] += red[threadIdx.x + st];
    __syncthreads();
  }
  float invs = 1.0f / red[0];
  unsigned short* pr = P + row * Wn;
#pragma unroll
  for (int i = 0; i < 4; ++i) pr[threadIdx.x + i * 256] = f2bf(v[i] * invs);
}

// ---------------- batched WMMA GEMM with async-LDS double-buffered tiles ----------------
// C[m][n] = alpha * sum_k A[m][k]*Bt[n][k] (+ C if accum). A,Bt bf16 row-major.
// Block: 256 threads = 8 waves (2x4), block tile 64(M) x 128(N), K step 32.
// Each wave computes a 2x2 grid of 16x16 WMMA tiles (32x32).
// LDS pitch 40 elems (80B) to avoid bank conflicts; 16B alignment preserved.
constexpr int LDS_PITCH = 40;
constexpr int TILE_BYTES = (64 + 128) * LDS_PITCH * 2;   // one buffer: A(64x32) + B(128x32)
constexpr int BOFF_BYTES = 64 * LDS_PITCH * 2;           // B area offset inside a buffer

__global__ __launch_bounds__(256) void wmma_gemm_bf16_kernel(
    const unsigned short* __restrict__ A, const unsigned short* __restrict__ Bt,
    float* __restrict__ Cf, unsigned short* __restrict__ Cbf,
    int K, int lda, int ldb, int ldc,
    float alpha, int accum, int zdiv,
    long long sAo, long long sAi, long long sBo, long long sBi,
    long long sCo, long long sCi) {
  __shared__ unsigned short lds[2][(64 + 128) * LDS_PITCH];

  int z = blockIdx.z;
  int zo = z / zdiv, zi = z % zdiv;
  A  += (long long)zo * sAo + (long long)zi * sAi;
  Bt += (long long)zo * sBo + (long long)zi * sBi;
  long long cb = (long long)zo * sCo + (long long)zi * sCi;

  const int tid = threadIdx.x;
  const int wid = tid >> 5;
  const int lane = tid & 31;
  const int wm = wid >> 2;        // 0..1
  const int wn = wid & 3;         // 0..3
  const int half = lane >> 4;
  const int l16 = lane & 15;
  const int mblk = blockIdx.y * 64;
  const int nblk = blockIdx.x * 128;

  unsigned ldsbase = (unsigned)(unsigned long long)(&lds[0][0]);

  // async global -> LDS tile stage: 3 async b128 issues per wave per tile
  auto issueTile = [&](int kt, int buf) {
    int k0 = kt * 32;
    unsigned base = ldsbase + (unsigned)buf * (unsigned)TILE_BYTES;
    {   // A tile 64x32: row = tid>>2, 16B segment = tid&3
      int row = tid >> 2, seg = tid & 3;
      const unsigned short* g = A + (long long)(mblk + row) * lda + k0 + seg * 8;
      unsigned l = base + (unsigned)(row * (LDS_PITCH * 2) + seg * 16);
      asm volatile("global_load_async_to_lds_b128 %0, %1, off"
                   :: "v"(l), "v"(g) : "memory");
    }
    {   // B tile 128x32: row = tid>>1, segments {tid&1, 2+(tid&1)}
      int row = tid >> 1, s0 = tid & 1;
      const unsigned short* g0 = Bt + (long long)(nblk + row) * ldb + k0 + s0 * 8;
      unsigned l0 = base + (unsigned)BOFF_BYTES + (unsigned)(row * (LDS_PITCH * 2) + s0 * 16);
      asm volatile("global_load_async_to_lds_b128 %0, %1, off"
                   :: "v"(l0), "v"(g0) : "memory");
      const unsigned short* g1 = g0 + 16;   // +2 segments
      unsigned l1 = l0 + 32;
      asm volatile("global_load_async_to_lds_b128 %0, %1, off"
                   :: "v"(l1), "v"(g1) : "memory");
    }
  };

  f32x8 acc00 = {}, acc01 = {}, acc10 = {}, acc11 = {};

  const int KT = K >> 5;
  issueTile(0, 0);
#pragma unroll 1
  for (int kt = 0; kt < KT; ++kt) {
    int cur = kt & 1;
    if (kt + 1 < KT) {
      issueTile(kt + 1, cur ^ 1);
      asm volatile("s_wait_asynccnt 0x3" ::: "memory");  // previous tile's 3 done (in-order)
    } else {
      asm volatile("s_wait_asynccnt 0x0" ::: "memory");
    }
    __syncthreads();

    const unsigned short* As = &lds[cur][0];
    const unsigned short* Bs = &lds[cur][64 * LDS_PITCH];

    FragU a0, a1, b0, b1;
    int ar0 = wm * 32 + l16, ar1 = ar0 + 16;
    a0.q[0] = *(const u32x4*)(As + ar0 * LDS_PITCH + half * 8);
    a0.q[1] = *(const u32x4*)(As + ar0 * LDS_PITCH + 16 + half * 8);
    a1.q[0] = *(const u32x4*)(As + ar1 * LDS_PITCH + half * 8);
    a1.q[1] = *(const u32x4*)(As + ar1 * LDS_PITCH + 16 + half * 8);
    int br0 = wn * 32 + l16, br1 = br0 + 16;
    b0.q[0] = *(const u32x4*)(Bs + br0 * LDS_PITCH + half * 8);
    b0.q[1] = *(const u32x4*)(Bs + br0 * LDS_PITCH + 16 + half * 8);
    b1.q[0] = *(const u32x4*)(Bs + br1 * LDS_PITCH + half * 8);
    b1.q[1] = *(const u32x4*)(Bs + br1 * LDS_PITCH + 16 + half * 8);

    acc00 = __builtin_amdgcn_wmma_f32_16x16x32_bf16(false, a0.v, false, b0.v, (short)0, acc00, false, false);
    acc01 = __builtin_amdgcn_wmma_f32_16x16x32_bf16(false, a0.v, false, b1.v, (short)0, acc01, false, false);
    acc10 = __builtin_amdgcn_wmma_f32_16x16x32_bf16(false, a1.v, false, b0.v, (short)0, acc10, false, false);
    acc11 = __builtin_amdgcn_wmma_f32_16x16x32_bf16(false, a1.v, false, b1.v, (short)0, acc11, false, false);

    __syncthreads();  // all waves done reading before this buffer is overwritten
  }

  // epilogue: 4 tiles of 16x16
  const f32x8* accs[4] = {&acc00, &acc01, &acc10, &acc11};
#pragma unroll
  for (int i = 0; i < 2; ++i) {
#pragma unroll
    for (int j = 0; j < 2; ++j) {
      const f32x8& a = *accs[i * 2 + j];
      int n = nblk + wn * 32 + j * 16 + l16;
#pragma unroll
      for (int rr = 0; rr < 8; ++rr) {
        int m = mblk + wm * 32 + i * 16 + rr + half * 8;
        long long idx = cb + (long long)m * ldc + n;
        float val = alpha * a[rr];
        if (Cf) {
          if (accum) val += Cf[idx];
          Cf[idx] = val;
        } else {
          Cbf[idx] = f2bf(val);
        }
      }
    }
  }
}

// ---------------- host orchestration ----------------
extern "C" void kernel_launch(void* const* d_in, const int* in_sizes, int n_in,
                              void* d_out, int out_size, void* d_ws, size_t ws_size,
                              hipStream_t stream) {
  (void)in_sizes; (void)n_in; (void)out_size; (void)ws_size;
  const float* x = (const float*)d_in[0];

  // workspace bump allocator (256B aligned)
  char* wsp = (char*)d_ws;
  size_t off = 0;
  auto take = [&](size_t bytes) -> void* {
    off = (off + 255) & ~(size_t)255;
    void* p = wsp + off;
    off += bytes;
    return p;
  };
  const size_t HSZ = (size_t)Bn * Cn * FW * 4;  // 33.5 MB
  float* hb0  = (float*)take(HSZ);
  float* hb1  = (float*)take(HSZ);
  float* h1   = (float*)take(HSZ);
  float* zbuf = (float*)take(HSZ);
  float* qkbuf = (float*)take((size_t)Bn * Hn * WW * 4);
  unsigned short* xt     = (unsigned short*)take((size_t)Bn * FW * 2);
  unsigned short* qlinbf = (unsigned short*)take((size_t)FW * 2);
  unsigned short* klinbf = (unsigned short*)take((size_t)FW * 2);
  unsigned short* vlinbf = (unsigned short*)take((size_t)FW * 2);
  unsigned short* opwbf  = (unsigned short*)take((size_t)FW * 2);
  // union region: attention scratch (alive before FFN) ∪ FFN mid buffer
  size_t region_off = (off + 255) & ~(size_t)255;
  size_t roff = region_off;
  auto rtake = [&](size_t bytes) -> void* {
    roff = (roff + 255) & ~(size_t)255;
    void* p = wsp + roff;
    roff += bytes;
    return p;
  };
  float* Qlin = (float*)rtake((size_t)Bn * FW * 4);
  float* Klin = (float*)rtake((size_t)Bn * FW * 4);
  float* Vlin = (float*)rtake((size_t)Bn * FW * 4);
  unsigned short* Vbf = (unsigned short*)rtake((size_t)Bn * FW * 2);
  unsigned short* Qh  = (unsigned short*)rtake((size_t)Bn * FW * 2);
  unsigned short* Kh  = (unsigned short*)rtake((size_t)Bn * FW * 2);
  unsigned short* Pbf = (unsigned short*)rtake((size_t)Bn * Hn * WW * 2);
  unsigned short* abt = (unsigned short*)rtake((size_t)Bn * FW * 2);
  float* olin = (float*)rtake((size_t)Bn * FW * 4);
  float* mid  = (float*)(wsp + region_off);  // [B,16,F,W] aliases attention scratch

  auto gemm = [&](const void* Abf, const void* Btbf, float* Cf, unsigned short* Cbf,
                  int M, int N, int K, int lda, int ldb, int ldc,
                  float alpha, int accum, int nb, int zdiv,
                  long long sAo, long long sAi, long long sBo, long long sBi,
                  long long sCo, long long sCi) {
    dim3 g(N / 128, M / 64, nb);
    wmma_gemm_bf16_kernel<<<g, 256, 0, stream>>>(
        (const unsigned short*)Abf, (const unsigned short*)Btbf, Cf, Cbf,
        K, lda, ldb, ldc, alpha, accum, zdiv, sAo, sAi, sBo, sBi, sCo, sCi);
  };

  const float* hprev = nullptr;
  float* hb[2] = {hb0, hb1};

  for (int l = 0; l < Ln; ++l) {
    const float* const* pp = (const float* const*)(d_in + 1 + l * 21);
    const float* norm1_w = pp[0];  const float* norm1_b = pp[1];
    const float* q_conv_w = pp[2]; const float* q_conv_b = pp[3]; const float* q_lin_w = pp[4];
    const float* k_conv_w = pp[5]; const float* k_conv_b = pp[6]; const float* k_lin_w = pp[7];
    const float* v_conv_w = pp[8]; const float* v_conv_b = pp[9]; const float* v_lin_w = pp[10];
    const float* o_lin_pw = pp[11]; const float* o_lin_dw = pp[12];
    const float* o_proj_w = pp[13]; const float* o_proj_b = pp[14];
    const float* norm2_w = pp[15]; const float* norm2_b = pp[16];
    const float* c1_w = pp[17]; const float* c1_b = pp[18];
    const float* c2_w = pp[19]; const float* c2_b = pp[20];

    const float* hin = (l == 0) ? x : hprev;
    float* hout = (l == Ln - 1) ? (float*)d_out : hb[l & 1];

    // convert linear weights f32 -> bf16 (F x F each)
    f2bf_kernel<<<4096, 256, 0, stream>>>(q_lin_w, qlinbf, FW);
    f2bf_kernel<<<4096, 256, 0, stream>>>(k_lin_w, klinbf, FW);
    f2bf_kernel<<<4096, 256, 0, stream>>>(v_lin_w, vlinbf, FW);
    f2bf_kernel<<<4096, 256, 0, stream>>>(o_lin_pw, opwbf, FW);

    // ---- Q path: conv -> xt (transposed bf16); Qlin = Wq * conv; rotary+pack
    conv_qkv_bt_kernel<<<8192, 256, 0, stream>>>(hin, q_conv_w, q_conv_b, xt);
    gemm(qlinbf, xt, Qlin, nullptr, Fn, Wn, Fn, Fn, Fn, Wn, 1.0f, 0, Bn, 1,
         0, 0, (long long)Wn * Fn, 0, (long long)Fn * Wn, 0);
    rotary_pack_kernel<<<8192, 256, 0, stream>>>(Qlin, Qh);
    // ---- K path
    conv_qkv_bt_kernel<<<8192, 256, 0, stream>>>(hin, k_conv_w, k_conv_b, xt);
    gemm(klinbf, xt, Klin, nullptr, Fn, Wn, Fn, Fn, Fn, Wn, 1.0f, 0, Bn, 1,
         0, 0, (long long)Wn * Fn, 0, (long long)Fn * Wn, 0);
    rotary_pack_kernel<<<8192, 256, 0, stream>>>(Klin, Kh);
    // ---- V path (no rotary; keep [g][w] layout which is B-transposed for P·V)
    conv_qkv_bt_kernel<<<8192, 256, 0, stream>>>(hin, v_conv_w, v_conv_b, xt);
    gemm(vlinbf, xt, Vlin, nullptr, Fn, Wn, Fn, Fn, Fn, Wn, 1.0f, 0, Bn, 1,
         0, 0, (long long)Wn * Fn, 0, (long long)Fn * Wn, 0);
    f2bf_kernel<<<8192, 256, 0, stream>>>(Vlin, Vbf, (long long)Bn * FW);

    // ---- qk = (l>0 ? qk : 0) + Q K^T / sqrt(F); per (b,h) batch of 8
    gemm(Qh, Kh, qkbuf, nullptr, Wn, Wn, HD, HD, HD, Wn, 1.0f / 32.0f, (l > 0) ? 1 : 0,
         Bn * Hn, Hn,
         (long long)Hn * Wn * HD, (long long)Wn * HD,
         (long long)Hn * Wn * HD, (long long)Wn * HD,
         (long long)Hn * WW, (long long)WW);
    softmax_kernel<<<Bn * Hn * Wn, 256, 0, stream>>>(qkbuf, Pbf);
    // ---- A = P V, stored transposed: abt[b][w][h*HD+dd]
    gemm(Pbf, Vbf, nullptr, abt, Wn, HD, Wn, Wn, Wn, Fn, 1.0f, 0, Bn * Hn, Hn,
         (long long)Hn * WW, (long long)WW,
         (long long)Fn * Wn, (long long)HD * Wn,
         (long long)Wn * Fn, (long long)HD);
    // ---- o_lin positionwise: olin[b][g][w] = Opw * abt[b]^T
    gemm(opwbf, abt, olin, nullptr, Fn, Wn, Fn, Fn, Fn, Wn, 1.0f, 0, Bn, 1,
         0, 0, (long long)Wn * Fn, 0, (long long)Fn * Wn, 0);
    // ---- o_proj conv over concat[x, dw*olin]
    oproj_conv_kernel<<<32768, 256, 0, stream>>>(hin, olin, o_lin_dw, o_proj_w, o_proj_b, zbuf);
    // ---- h1 = LN(x + z)
    add_ln_kernel<<<Bn * Cn * Wn, 256, 0, stream>>>(hin, zbuf, norm1_w, norm1_b, h1);
    // ---- FFN: c1+gelu -> mid; c2 -> zbuf; h2 = LN(h1 + z2)
    conv3x3_kernel<<<131072, 256, 0, stream>>>(h1, c1_w, c1_b, mid, Cn, CE, 1);
    conv3x3_kernel<<<32768, 256, 0, stream>>>(mid, c2_w, c2_b, zbuf, CE, Cn, 0);
    add_ln_kernel<<<Bn * Cn * Wn, 256, 0, stream>>>(h1, zbuf, norm2_w, norm2_b, hout);

    hprev = hout;
  }
}
